// EnhancedQuantumInspiredLSTM_16767552324223
// MI455X (gfx1250) — matrix operand order repaired and checked by reading, not compiled, not run on hardware
//
#include <hip/hip_runtime.h>
#include <math.h>

// Problem constants (from reference)
#define I_   128
#define H_   512
#define B_   64
#define S_   512
#define M_   (B_ * S_)      // 32768 rows (time-major: m = s*B + b)
#define G4_  (4 * H_)       // 2048 gate columns
#define NWG  16             // workgroups in recurrence kernel

typedef __bf16 bf16;
typedef __attribute__((ext_vector_type(16))) __bf16 v16bf;
typedef __attribute__((ext_vector_type(8)))  __bf16 v8bf;
typedef __attribute__((ext_vector_type(8)))  float  v8f;

// ---------------- bf16 helpers (explicit bit manipulation, no header deps) ----
__device__ inline bf16 f2bf(float f) {
  union { float f; unsigned u; } x; x.f = f;
  unsigned r = x.u + 0x7FFFu + ((x.u >> 16) & 1u);   // round-to-nearest-even
  union { unsigned short u; bf16 b; } o; o.u = (unsigned short)(r >> 16);
  return o.b;
}
__device__ inline float bf2f(bf16 b) {
  union { bf16 b; unsigned short u; } i; i.b = b;
  union { unsigned u; float f; } o; o.u = ((unsigned)i.u) << 16;
  return o.f;
}

// ---------------- CDNA5 async global->LDS copy (ASYNCcnt path) ----------------
// GLOBAL_LOAD_ASYNC_TO_LDS_B128: VDST = per-lane LDS byte address, VADDR = per-
// lane 64-bit global address. Generic pointers to __shared__ carry the LDS byte
// offset in their low 32 bits (AMDGPU addrspacecast convention / ISA aperture
// truncation), so we pass that directly.
__device__ inline void async_ld_b128(void* lds_ptr, const void* gaddr) {
  unsigned lds = (unsigned)(size_t)lds_ptr;
  asm volatile("global_load_async_to_lds_b128 %0, %1, off"
               :: "v"(lds), "v"(gaddr) : "memory");
}
__device__ inline void wait_async0() {
  asm volatile("s_wait_asynccnt 0x0" ::: "memory");
}

// ---------------- WMMA core ---------------------------------------------------
__device__ inline v8f wmma_bf16(v16bf a, v16bf b, v8f c) {
  // D = A(16x32 bf16) * B(32x16 bf16) + C(16x16 f32)
  return __builtin_amdgcn_wmma_f32_16x16x32_bf16(
      /*neg_a=*/false, a, /*neg_b=*/false, b,
      /*c_mod=*/(short)0, c, /*reuse_a=*/false, /*reuse_b=*/false);
}

// 16-bit A-matrix 16x32 fragment from row-major A (lda in elements).
// ISA layout: lanes 0-15 row M, elems 0..7 = K0..7, 8..15 = K16..23;
// lanes 16-31 same rows, elems 0..7 = K8..15, 8..15 = K24..31.
__device__ inline v16bf load_a_frag(const bf16* A, int lda, int m0, int k0, int lane) {
  const bf16* p = A + (size_t)(m0 + (lane & 15)) * lda + (k0 + ((lane >> 4) << 3));
  v8bf lo = *(const v8bf*)p;          // 16B load
  v8bf hi = *(const v8bf*)(p + 16);   // 16B load
  v16bf a;
#pragma unroll
  for (int i = 0; i < 8; ++i) { a[i] = lo[i]; a[i + 8] = hi[i]; }
  return a;
}

// 16-bit B-matrix 32x16 fragment from W stored (N,K) row-major (ldk in elems).
// lanes 0-15: col N=n0+lane, elems 0..7 = K0..7, 8..15 = K16..23; lanes 16-31: K8..15, K24..31.
__device__ inline v16bf load_b_frag(const bf16* W, int ldk, int n0, int k0, int lane) {
  const bf16* p = W + (size_t)(n0 + (lane & 15)) * ldk + (k0 + ((lane >> 4) << 3));
  v8bf lo = *(const v8bf*)p;
  v8bf hi = *(const v8bf*)(p + 16);
  v16bf b;
#pragma unroll
  for (int i = 0; i < 8; ++i) { b[i] = lo[i]; b[i + 8] = hi[i]; }
  return b;
}

// ---------------- prep kernels ------------------------------------------------
// cos/sin weights, transposed to (H, I) row-major bf16 so they load as B-frags.
__global__ __launch_bounds__(256) void k_prep_qw(
    const float* __restrict__ th, const float* __restrict__ ph,
    const float* __restrict__ thn, const float* __restrict__ phn,
    bf16* __restrict__ cosT, bf16* __restrict__ sinT) {
  int idx = blockIdx.x * 256 + threadIdx.x;          // over I_*H_
  if (idx >= I_ * H_) return;
  int i = idx >> 9, h = idx & (H_ - 1);
  cosT[(h << 7) + i] = f2bf(cosf(th[idx] + thn[idx]));
  sinT[(h << 7) + i] = f2bf(sinf(ph[idx] + phn[idx]));
}

// x (B,S,I) f32 -> time-major (s*B+b, I) bf16
__global__ __launch_bounds__(256) void k_prep_x(const float* __restrict__ x,
                                                bf16* __restrict__ xb) {
  int idx = blockIdx.x * 256 + threadIdx.x;          // over B_*S_*I_ = 2^22
  if (idx >= B_ * S_ * I_) return;
  int i = idx & (I_ - 1);
  int s = (idx >> 7) & (S_ - 1);
  int b = idx >> 16;
  xb[(((size_t)s * B_ + b) << 7) + i] = f2bf(x[idx]);
}

__global__ __launch_bounds__(256) void k_cast_bf16(const float* __restrict__ src,
                                                   bf16* __restrict__ dst, int n) {
  int idx = blockIdx.x * 256 + threadIdx.x;
  if (idx < n) dst[idx] = f2bf(src[idx]);
}

__global__ __launch_bounds__(256) void k_bias_sum(const float* __restrict__ a,
                                                  const float* __restrict__ b,
                                                  float* __restrict__ o, int n) {
  int idx = blockIdx.x * 256 + threadIdx.x;
  if (idx < n) o[idx] = a[idx] + b[idx];
}

// ---------------- quantum layer: q = sqrt((xC)^2 + (xS)^2) --------------------
// One 16x16 tile per wave. M=32768 (2048 mtiles), N=512 (32 ntiles), K=128.
__global__ __launch_bounds__(256) void k_qgemm(const bf16* __restrict__ xb,
                                               const bf16* __restrict__ cosT,
                                               const bf16* __restrict__ sinT,
                                               bf16* __restrict__ qb) {
  int wave = threadIdx.x >> 5, lane = threadIdx.x & 31;
  int t = blockIdx.x * 8 + wave;                     // 65536 tiles
  int m0 = (t & 2047) << 4;
  int n0 = (t >> 11) << 4;
  v8f ar, ai;
#pragma unroll
  for (int r = 0; r < 8; ++r) { ar[r] = 0.f; ai[r] = 0.f; }
#pragma unroll
  for (int k0 = 0; k0 < I_; k0 += 32) {
    v16bf a = load_a_frag(xb, I_, m0, k0, lane);
    ar = wmma_bf16(a, load_b_frag(cosT, I_, n0, k0, lane), ar);
    ai = wmma_bf16(a, load_b_frag(sinT, I_, n0, k0, lane), ai);
  }
  int col = n0 + (lane & 15);
  int rbase = m0 + ((lane >> 4) << 3);
#pragma unroll
  for (int r = 0; r < 8; ++r) {
    float q = sqrtf(ar[r] * ar[r] + ai[r] * ai[r]);
    qb[(size_t)(rbase + r) * H_ + col] = f2bf(q);
  }
}

// ---------------- x_proj GEMM: (M x H) * (H x 4H)^T + bias -> f32 -------------
__global__ __launch_bounds__(256) void k_xproj(const bf16* __restrict__ A,
                                               const bf16* __restrict__ Wih,
                                               const float* __restrict__ bsum,
                                               float* __restrict__ xp) {
  int wave = threadIdx.x >> 5, lane = threadIdx.x & 31;
  int t = blockIdx.x * 8 + wave;                     // 262144 tiles
  int m0 = (t & 2047) << 4;
  int n0 = (t >> 11) << 4;
  int col = n0 + (lane & 15);
  float bb = bsum[col];
  v8f acc;
#pragma unroll
  for (int r = 0; r < 8; ++r) acc[r] = bb;
#pragma unroll
  for (int k0 = 0; k0 < H_; k0 += 32)
    acc = wmma_bf16(load_a_frag(A, H_, m0, k0, lane),
                    load_b_frag(Wih, H_, n0, k0, lane), acc);
  int rbase = m0 + ((lane >> 4) << 3);
#pragma unroll
  for (int r = 0; r < 8; ++r)
    xp[(size_t)(rbase + r) * G4_ + col] = acc[r];
}

// ---------------- persistent LSTM recurrence ----------------------------------
// 16 WGs x 512 threads. WG `wg` owns H-columns [wg*32, wg*32+32): gate slab of
// 128 rows of W_hh held resident in LDS; h (64x512 bf16) pulled from L2 each
// step via GLOBAL_LOAD_ASYNC_TO_LDS_B128, overlapped with x_proj acc loads.
__global__ __launch_bounds__(512) void k_recur(const float* __restrict__ xp,
                                               const bf16* __restrict__ Whh,
                                               bf16* __restrict__ hbuf,
                                               bf16* __restrict__ hs,
                                               int* __restrict__ syncp) {
  __shared__ __align__(16) bf16  sW[128 * H_];   // 128 KB: W_hh slab (N,K)
  __shared__ __align__(16) bf16  sH[B_ * H_];    //  64 KB: current h
  __shared__ __align__(16) float sG[B_ * 128];   //  32 KB: pre-activation gates
  int tid = threadIdx.x, wg = blockIdx.x;
  int lane = tid & 31, wave = tid >> 5;

  // Stage W_hh slab once (async DMA to LDS):
  // local row nl -> global gate row (nl/32)*512 + wg*32 + nl%32
  for (int idx = tid; idx < 128 * H_ / 8; idx += 512) {
    int nl = idx >> 6;
    int kc = (idx & 63) << 3;
    int grow = ((nl >> 5) << 9) + (wg << 5) + (nl & 31);
    async_ld_b128(sW + nl * H_ + kc, Whh + (size_t)grow * H_ + kc);
  }
  float cc[4] = {0.f, 0.f, 0.f, 0.f};             // cell state, 4 elems/thread
  wait_async0();
  __syncthreads();

  for (int s = 0; s < S_; ++s) {
    // issue async h exchange: global hbuf -> LDS (written by all WGs last step,
    // visibility guaranteed by the device barrier + threadfence below)
    for (int idx = tid; idx < B_ * H_ / 8; idx += 512)
      async_ld_b128((v8bf*)sH + idx, (const v8bf*)hbuf + idx);

    // overlap the in-flight async copies with x_proj accumulator loads
    float xacc[2][8];
#pragma unroll
    for (int tt = 0; tt < 2; ++tt) {
      int tindex = wave * 2 + tt;                 // 0..31
      int coll = ((tindex & 7) << 4) + (lane & 15);
      int gcol = ((coll >> 5) << 9) + (wg << 5) + (coll & 31);
      int rbase = ((tindex >> 3) << 4) + ((lane >> 4) << 3);
#pragma unroll
      for (int r = 0; r < 8; ++r) {
        size_t ofs = (size_t)(s * B_ + rbase + r) * G4_ + gcol;
        xacc[tt][r] = xp[ofs];
        __builtin_prefetch(xp + ofs + (size_t)B_ * G4_, 0, 0);  // next step
      }
    }
    wait_async0();
    __syncthreads();

    // gates = x_proj[s] + h @ Whh_slab^T : 32 tiles, 2 per wave, K=512
#pragma unroll
    for (int tt = 0; tt < 2; ++tt) {
      int tindex = wave * 2 + tt;
      int m0  = (tindex >> 3) << 4;               // batch rows
      int n0l = (tindex & 7) << 4;                // local gate cols
      int coll = n0l + (lane & 15);
      int rbase = m0 + ((lane >> 4) << 3);
      v8f acc;
#pragma unroll
      for (int r = 0; r < 8; ++r) acc[r] = xacc[tt][r];
#pragma unroll
      for (int k0 = 0; k0 < H_; k0 += 32)
        acc = wmma_bf16(load_a_frag(sH, H_, m0, k0, lane),
                        load_b_frag(sW, H_, n0l, k0, lane), acc);
#pragma unroll
      for (int r = 0; r < 8; ++r)
        sG[(rbase + r) * 128 + coll] = acc[r];
    }
    __syncthreads();

    // nonlinearities + state update (64 rows x 32 cols = 2048 elems / 512 thr)
#pragma unroll
    for (int i = 0; i < 4; ++i) {
      int e = i * 512 + tid;
      int b = e >> 5, hc = e & 31;
      float gi = sG[b * 128 + hc];
      float gf = sG[b * 128 + 32 + hc];
      float gg = sG[b * 128 + 64 + hc];
      float go = sG[b * 128 + 96 + hc];
      gi = 1.f / (1.f + __expf(-gi));
      gf = 1.f / (1.f + __expf(-gf));
      go = 1.f / (1.f + __expf(-go));
      gg = tanhf(gg);
      float c = gf * cc[i] + gi * gg;
      cc[i] = c;
      float h = go * tanhf(c);
      bf16 hb = f2bf(h);
      int col = (wg << 5) + hc;
      hbuf[b * H_ + col] = hb;                              // exchange buffer
      hs[((size_t)s * B_ + b) * H_ + col] = hb;             // layer output seq
    }

    // device-wide ticket barrier (monotonic counter, phase via division)
    __threadfence();
    __syncthreads();
    if (tid == 0) {
      int ticket = atomicAdd(syncp, 1);
      int target = ((ticket / NWG) + 1) * NWG;
      while (atomicAdd(syncp, 0) < target) { __builtin_amdgcn_s_sleep(1); }
    }
    __syncthreads();
  }
}

// ---------------- FC head: relu(last @ fc1^T + b1) @ fc2^T + b2 ---------------
__global__ __launch_bounds__(256) void k_head(const bf16* __restrict__ hs,
                                              const bf16* __restrict__ fc1w,
                                              const float* __restrict__ fc1b,
                                              const float* __restrict__ fc2w,
                                              const float* __restrict__ fc2b,
                                              float* __restrict__ out) {
  __shared__ float sLast[H_];
  __shared__ float sH1[H_];
  __shared__ float sRed[256];
  int b = blockIdx.x, tid = threadIdx.x;
  const bf16* row = hs + ((size_t)(S_ - 1) * B_ + b) * H_;
  for (int j = tid; j < H_; j += 256) sLast[j] = bf2f(row[j]);
  __syncthreads();
  for (int j = tid; j < H_; j += 256) {
    const bf16* w = fc1w + (size_t)j * H_;
    float acc = fc1b[j];
    for (int k = 0; k < H_; ++k) acc += sLast[k] * bf2f(w[k]);
    sH1[j] = fmaxf(acc, 0.f);
  }
  __syncthreads();
  float p = 0.f;
  for (int j = tid; j < H_; j += 256) p += sH1[j] * fc2w[j];
  sRed[tid] = p;
  __syncthreads();
  for (int off = 128; off > 0; off >>= 1) {
    if (tid < off) sRed[tid] += sRed[tid + off];
    __syncthreads();
  }
  if (tid == 0) out[b] = sRed[0] + fc2b[0];
}

// ---------------- host orchestration ------------------------------------------
extern "C" void kernel_launch(void* const* d_in, const int* in_sizes, int n_in,
                              void* d_out, int out_size, void* d_ws, size_t ws_size,
                              hipStream_t stream) {
  (void)in_sizes; (void)n_in; (void)out_size; (void)ws_size;
  const float* x      = (const float*)d_in[0];
  const float* theta  = (const float*)d_in[1];
  const float* phi    = (const float*)d_in[2];
  const float* thn    = (const float*)d_in[3];
  const float* phn    = (const float*)d_in[4];
  const float* W_ih   = (const float*)d_in[5];
  const float* W_hh   = (const float*)d_in[6];
  const float* b_ih   = (const float*)d_in[7];
  const float* b_hh   = (const float*)d_in[8];
  const float* fc1_w  = (const float*)d_in[9];
  const float* fc1_b  = (const float*)d_in[10];
  const float* fc2_w  = (const float*)d_in[11];
  const float* fc2_b  = (const float*)d_in[12];

  char* wsp = (char*)d_ws;
  size_t off = 0;
  auto carve = [&](size_t bytes) -> void* {
    void* p = wsp + off;
    off += (bytes + 255) & ~(size_t)255;
    return p;
  };
  bf16*  xb    = (bf16*) carve((size_t)M_ * I_ * 2);        //   8 MB
  bf16*  cosT  = (bf16*) carve((size_t)H_ * I_ * 2);
  bf16*  sinT  = (bf16*) carve((size_t)H_ * I_ * 2);
  bf16*  wihb  = (bf16*) carve((size_t)2 * G4_ * H_ * 2);   //   4 MB
  bf16*  whhb  = (bf16*) carve((size_t)2 * G4_ * H_ * 2);   //   4 MB
  float* bsum  = (float*)carve((size_t)2 * G4_ * 4);
  bf16*  fc1wb = (bf16*) carve((size_t)H_ * H_ * 2);
  bf16*  qb    = (bf16*) carve((size_t)M_ * H_ * 2);        //  32 MB
  float* xpj   = (float*)carve((size_t)M_ * G4_ * 4);       // 256 MB (reused)
  bf16*  hsb0  = (bf16*) carve((size_t)M_ * H_ * 2);        //  32 MB
  bf16*  hsb1  = (bf16*) carve((size_t)M_ * H_ * 2);        //  32 MB
  bf16*  hbuf  = (bf16*) carve((size_t)B_ * H_ * 2);
  int*   syncp = (int*)  carve(256);

  // prep
  k_prep_qw<<<(I_ * H_ + 255) / 256, 256, 0, stream>>>(theta, phi, thn, phn, cosT, sinT);
  k_prep_x<<<(B_ * S_ * I_ + 255) / 256, 256, 0, stream>>>(x, xb);
  k_cast_bf16<<<(2 * G4_ * H_ + 255) / 256, 256, 0, stream>>>(W_ih, wihb, 2 * G4_ * H_);
  k_cast_bf16<<<(2 * G4_ * H_ + 255) / 256, 256, 0, stream>>>(W_hh, whhb, 2 * G4_ * H_);
  k_cast_bf16<<<(H_ * H_ + 255) / 256, 256, 0, stream>>>(fc1_w, fc1wb, H_ * H_);
  k_bias_sum<<<(2 * G4_ + 255) / 256, 256, 0, stream>>>(b_ih, b_hh, bsum, 2 * G4_);

  // quantum layer GEMMs (fused real/imag/magnitude)
  k_qgemm<<<(M_ / 16) * (H_ / 16) / 8, 256, 0, stream>>>(xb, cosT, sinT, qb);

  // two LSTM layers: wide x_proj GEMM then persistent recurrence
  for (int l = 0; l < 2; ++l) {
    const bf16* Ain = (l == 0) ? qb : hsb0;
    bf16* hsOut = (l == 0) ? hsb0 : hsb1;
    k_xproj<<<(M_ / 16) * (G4_ / 16) / 8, 256, 0, stream>>>(
        Ain, wihb + (size_t)l * G4_ * H_, bsum + (size_t)l * G4_, xpj);
    hipMemsetAsync(hbuf, 0, (size_t)B_ * H_ * 2, stream);
    hipMemsetAsync(syncp, 0, 256, stream);
    k_recur<<<NWG, 512, 0, stream>>>(xpj, whhb + (size_t)l * G4_ * H_, hbuf, hsOut, syncp);
  }

  // head
  k_head<<<B_, 256, 0, stream>>>(hsb1, fc1wb, fc1_b, fc2_w, fc2_b, (float*)d_out);
}
